// StructureUpdate_50964081935579
// MI455X (gfx1250) — matrix operand order repaired and checked by reading, not compiled, not run on hardware
//
#include <hip/hip_runtime.h>

typedef __attribute__((ext_vector_type(16))) _Float16 v16h;
typedef __attribute__((ext_vector_type(8)))  float    v8f;
typedef __attribute__((ext_vector_type(2)))  float    vf2;

#define NN 1024
#define INV_SQRT3f 0.5773502691896258f
#define WCf 0.11785113019775793f   // sqrt(2/PQ)/6, PQ=4

// ---------------------------------------------------------------------------
// Generic batched WMMA GEMM: C[b][m][n] = sum_k A[b][m][k]*B[b][k][n]
//   (+ bias[n]) (+ add[b][m][n]) (relu). One wave per 16x16 tile.
// grid = (Ntiles, Mtiles, batch), block = 32.
// K must be a multiple of 32. Ragged M/N handled by clamping A-row / B-col
// indices; stores are UNCONDITIONAL — caller guarantees that any tile padding
// lands in memory that is overwritten later or never read.
// Fragment layouts (gfx1250 wave32, f16):
//   A 16x32: lane l: m=l&15, hi=l>>4; reg r: k=(r>>2)*16 + hi*8 + (r&3)*2
//   B 32x16: lane l: n=l&15, hi=l>>4; reg r: k=(r>>2)*16 + hi*8 + (r&3)*2
//   C 16x16: lane l: n=l&15, hi=l>>4; reg r: m=hi*8+r
// ---------------------------------------------------------------------------
__global__ void gemm_wmma_kernel(const float* __restrict__ A, long lda, long sA,
                                 const float* __restrict__ B, long ldb, long sB,
                                 float* __restrict__ C, long ldc, long sC,
                                 const float* __restrict__ bias,
                                 const float* __restrict__ add, long ldadd, long sAdd,
                                 int mmax, int nmax, int K, int do_relu) {
  int nt = blockIdx.x, mt = blockIdx.y, b = blockIdx.z;
  int lane = threadIdx.x & 31;
  int lo = lane & 15, hi = lane >> 4;
  int ma = mt * 16 + lo; if (ma > mmax) ma = mmax;
  int na = nt * 16 + lo; if (na > nmax) na = nmax;
  const float* arow = A + (long)b * sA + (long)ma * lda;   // lane's A row
  const float* bcol = B + (long)b * sB + na;               // lane's B column
  v8f acc = {};
  for (int k0 = 0; k0 < K; k0 += 32) {
    if (k0 + 32 < K) {
      __builtin_prefetch(arow + k0 + 32, 0, 1);
      __builtin_prefetch(bcol + (long)(k0 + 32) * ldb, 0, 1);
    }
    v16h af;
#pragma unroll
    for (int r = 0; r < 8; ++r) {
      int k = k0 + (r >> 2) * 16 + hi * 8 + (r & 3) * 2;
      vf2 x = *(const vf2*)(arow + k);                     // 8B aligned
      af[2 * r]     = (_Float16)x.x;
      af[2 * r + 1] = (_Float16)x.y;
    }
    v16h bf;
#pragma unroll
    for (int r = 0; r < 8; ++r) {
      long k = k0 + (r >> 2) * 16 + hi * 8 + (r & 3) * 2;
      bf[2 * r]     = (_Float16)bcol[k * ldb];
      bf[2 * r + 1] = (_Float16)bcol[(k + 1) * ldb];
    }
    acc = __builtin_amdgcn_wmma_f32_16x16x32_f16(false, af, false, bf, (short)0, acc,
                                                 false, false);
  }
  // --- epilogue: branch-hoisted, unconditional stores
  int n = nt * 16 + lo;
  float bv = bias ? bias[n] : 0.f;
  float av[8];
  if (add) {
    const float* ad = add + (long)b * sAdd + (long)(mt * 16 + hi * 8) * ldadd + n;
#pragma unroll
    for (int r = 0; r < 8; ++r) av[r] = ad[(long)r * ldadd];
  } else {
#pragma unroll
    for (int r = 0; r < 8; ++r) av[r] = 0.f;
  }
  float* crow = C + (long)b * sC + (long)(mt * 16 + hi * 8) * ldc + n;
#pragma unroll
  for (int r = 0; r < 8; ++r) {
    float v = acc[r] + bv + av[r];
    if (do_relu) v = fmaxf(v, 0.f);
    crow[(long)r * ldc] = v;
  }
}

// ---------------------------------------------------------------------------
// Rigid point transforms: gq/gk (H*PQ=48 pts), gv (H*PV=96 pts), squared norms
// grid = N, block = 96
// ---------------------------------------------------------------------------
__global__ void pts_kernel(const float* __restrict__ vqk, const float* __restrict__ vv,
                           const float* __restrict__ R, const float* __restrict__ tr,
                           float* __restrict__ gq, float* __restrict__ gk,
                           float* __restrict__ gv, float* __restrict__ sqq,
                           float* __restrict__ sqk) {
  __shared__ float pq[48], pk[48];
  int n = blockIdx.x, t = threadIdx.x;
  float Rm[9];
#pragma unroll
  for (int i = 0; i < 9; ++i) Rm[i] = R[(long)n * 9 + i];
  float tv[3];
#pragma unroll
  for (int a = 0; a < 3; ++a) tv[a] = tr[(long)n * 3 + a];

  if (t < 48) {
    int h = t >> 2, p = t & 3;
    const float* vq = vqk + (long)n * 288 + h * 12 + p * 3;
    const float* vk = vqk + (long)n * 288 + 144 + h * 12 + p * 3;
    float ssq = 0.f, ssk = 0.f;
#pragma unroll
    for (int a = 0; a < 3; ++a) {
      float q = Rm[a * 3] * vq[0] + Rm[a * 3 + 1] * vq[1] + Rm[a * 3 + 2] * vq[2] + tv[a];
      float k = Rm[a * 3] * vk[0] + Rm[a * 3 + 1] * vk[1] + Rm[a * 3 + 2] * vk[2] + tv[a];
      gq[(long)n * 144 + h * 12 + p * 3 + a] = q;
      gk[(long)n * 144 + h * 12 + p * 3 + a] = k;
      ssq += q * q;
      ssk += k * k;
    }
    pq[t] = ssq;
    pk[t] = ssk;
  }
  {
    int h = t >> 3, p = t & 7;
    const float* v = vv + (long)n * 288 + h * 24 + p * 3;
#pragma unroll
    for (int a = 0; a < 3; ++a)
      gv[(long)n * 288 + h * 24 + p * 3 + a] =
          Rm[a * 3] * v[0] + Rm[a * 3 + 1] * v[1] + Rm[a * 3 + 2] * v[2] + tv[a];
  }
  __syncthreads();
  if (t < 12) {
    float s1 = 0.f, s2 = 0.f;
#pragma unroll
    for (int p = 0; p < 4; ++p) { s1 += pq[t * 4 + p]; s2 += pk[t * 4 + p]; }
    sqq[(long)n * 12 + t] = s1;
    sqk[(long)n * 12 + t] = s2;
  }
}

// Unconditional select-load for the padded K=32 fragment: scalar part (16),
// point part (12), zero pad (4). One load + one mul per element.
__device__ inline float sel_load(const float* __restrict__ sp,
                                 const float* __restrict__ gp, int k, float s_sc,
                                 float s_pt) {
  int kg = k - 16;
  if (kg > 11) kg = 0;                      // keep address in-bounds
  const float* p = (k < 16) ? (sp + k) : (gp + kg);
  float s = (k < 16) ? s_sc : ((k < 28) ? s_pt : 0.f);
  return *p * s;
}

// ---------------------------------------------------------------------------
// Fused attention logits: edge-bias WMMA (edge read once) + QK/point WMMA.
// grid = (64 j-tiles, 64 i-tiles), block = (32, 16) = 16 waves
// Phase 1: wave w computes eb for ij-pairs [w*16, w*16+16) x 12 heads -> LDS
// Phase 2: wave h (h<12) computes a 16x16 logits tile for head h (K=32 fused)
// ---------------------------------------------------------------------------
__global__ __launch_bounds__(512) void logits_kernel(
    const float* __restrict__ edge, const float* __restrict__ w_eb,
    const float* __restrict__ qkv, const float* __restrict__ gq,
    const float* __restrict__ gk, const float* __restrict__ sqq,
    const float* __restrict__ sqk, const float* __restrict__ pw,
    float* __restrict__ logits) {
  __shared__ float eb[256 * 16];
  int lane = threadIdx.x;
  int w = threadIdx.y;
  int hi = lane >> 4;
  int jt = blockIdx.x, it = blockIdx.y;

  // ---- phase 1: edge bias, M = 256 ij pairs, N = 12 heads, K = 64 channels
  {
    v8f acc = {};
    int mloc = w * 16 + (lane & 15);
    int gi = it * 16 + (mloc >> 4);
    int gj = jt * 16 + (mloc & 15);
    const float* arow = edge + ((long)gi * NN + gj) * 64;
    int nb = lane & 15; if (nb > 11) nb = 11;          // clamp B column
    const float* bcol = w_eb + nb;
    for (int k0 = 0; k0 < 64; k0 += 32) {
      v16h af;
#pragma unroll
      for (int r = 0; r < 8; ++r) {
        int k = k0 + (r >> 2) * 16 + hi * 8 + (r & 3) * 2;
        vf2 x = *(const vf2*)(arow + k);
        af[2 * r]     = (_Float16)x.x;
        af[2 * r + 1] = (_Float16)x.y;
      }
      v16h bf;
#pragma unroll
      for (int r = 0; r < 8; ++r) {
        int k = k0 + (r >> 2) * 16 + hi * 8 + (r & 3) * 2;
        bf[2 * r]     = (_Float16)bcol[k * 12];
        bf[2 * r + 1] = (_Float16)bcol[(k + 1) * 12];
      }
      acc = __builtin_amdgcn_wmma_f32_16x16x32_f16(false, af, false, bf, (short)0, acc,
                                                   false, false);
    }
    int nn = lane & 15;
#pragma unroll
    for (int r = 0; r < 8; ++r) {
      int m = w * 16 + hi * 8 + r;
      eb[m * 16 + nn] = acc[r];
    }
  }
  __syncthreads();

  // ---- phase 2: per-head logits; K = [16 scalar | 12 point | 4 zero]
  if (w < 12) {
    int h = w;
    float pwh = pw[h];
    float s_sc = INV_SQRT3f * 0.25f;              // 1/sqrt(HD)=1/4 folded
    float s_pt = 2.f * INV_SQRT3f * WCf * pwh;    // +2*WC*pw*cross folded
    float c_h  = INV_SQRT3f * WCf * pwh;
    v16h af, bf;
    {
      int i = it * 16 + (lane & 15);
      const float* sqp = qkv + (long)i * 576 + h * 16;        // sq
      const float* gqp = gq + (long)i * 144 + h * 12;
#pragma unroll
      for (int r = 0; r < 8; ++r) {
        int k = (r >> 2) * 16 + hi * 8 + (r & 3) * 2;
        af[2 * r]     = (_Float16)sel_load(sqp, gqp, k, s_sc, s_pt);
        af[2 * r + 1] = (_Float16)sel_load(sqp, gqp, k + 1, s_sc, s_pt);
      }
    }
    {
      int j = jt * 16 + (lane & 15);
      const float* skp = qkv + (long)j * 576 + 192 + h * 16;  // sk
      const float* gkp = gk + (long)j * 144 + h * 12;
#pragma unroll
      for (int r = 0; r < 8; ++r) {
        int k = (r >> 2) * 16 + hi * 8 + (r & 3) * 2;
        bf[2 * r]     = (_Float16)sel_load(skp, gkp, k, 1.f, 1.f);
        bf[2 * r + 1] = (_Float16)sel_load(skp, gkp, k + 1, 1.f, 1.f);
      }
    }
    v8f acc = {};
    acc = __builtin_amdgcn_wmma_f32_16x16x32_f16(false, af, false, bf, (short)0, acc,
                                                 false, false);
    int nn = lane & 15;
    int j = jt * 16 + nn;
    float skj = sqk[(long)j * 12 + h];
#pragma unroll
    for (int r = 0; r < 8; ++r) {
      int mloc = hi * 8 + r;
      int i = it * 16 + mloc;
      float v = acc[r] + INV_SQRT3f * eb[(mloc * 16 + nn) * 16 + h] -
                c_h * (sqq[(long)i * 12 + h] + skj);
      logits[(long)h * NN * NN + (long)i * NN + j] = v;
    }
  }
}

// ---------------------------------------------------------------------------
// Row softmax in place (row length 1024). grid = 12*1024, block = 256
// ---------------------------------------------------------------------------
__global__ void softmax_kernel(float* __restrict__ logits) {
  __shared__ float red[256];
  long row = blockIdx.x;
  float* p = logits + row * NN;
  int t = threadIdx.x;
  float v[4];
  float mx = -1e30f;
#pragma unroll
  for (int i = 0; i < 4; ++i) { v[i] = p[t + 256 * i]; mx = fmaxf(mx, v[i]); }
  red[t] = mx; __syncthreads();
  for (int s = 128; s > 0; s >>= 1) { if (t < s) red[t] = fmaxf(red[t], red[t + s]); __syncthreads(); }
  mx = red[0]; __syncthreads();
  float sum = 0.f;
#pragma unroll
  for (int i = 0; i < 4; ++i) { v[i] = expf(v[i] - mx); sum += v[i]; }
  red[t] = sum; __syncthreads();
  for (int s = 128; s > 0; s >>= 1) { if (t < s) red[t] += red[t + s]; __syncthreads(); }
  float inv = 1.f / red[0];
#pragma unroll
  for (int i = 0; i < 4; ++i) p[t + 256 * i] = v[i] * inv;
}

// ---------------------------------------------------------------------------
// Vector epilogue: loc = R^T (agg - t), norm; write into combined[:,960:1344]
// agg is padded to 32 floats per head. grid = N, block = 96
// ---------------------------------------------------------------------------
__global__ void vec_kernel(const float* __restrict__ agg, const float* __restrict__ R,
                           const float* __restrict__ tr, float* __restrict__ combined) {
  int n = blockIdx.x, t = threadIdx.x;
  int h = t >> 3, p = t & 7;
  const float* Rn = R + (long)n * 9;
  float x[3];
#pragma unroll
  for (int a = 0; a < 3; ++a)
    x[a] = agg[(long)n * 384 + h * 32 + p * 3 + a] - tr[(long)n * 3 + a];
  float loc[3];
#pragma unroll
  for (int a = 0; a < 3; ++a) loc[a] = Rn[a] * x[0] + Rn[3 + a] * x[1] + Rn[6 + a] * x[2];
  float nrm = sqrtf(loc[0] * loc[0] + loc[1] * loc[1] + loc[2] * loc[2]);
  float* o = combined + (long)n * 1344 + 960 + h * 32 + p * 4;
  o[0] = nrm; o[1] = loc[0]; o[2] = loc[1]; o[3] = loc[2];
}

// ---------------------------------------------------------------------------
// LayerNorm over width 128. grid = N, block = 128
// ---------------------------------------------------------------------------
__global__ void ln_kernel(const float* __restrict__ in, const float* __restrict__ g,
                          const float* __restrict__ b, float* __restrict__ out) {
  __shared__ float red[128];
  int n = blockIdx.x, t = threadIdx.x;
  float x = in[(long)n * 128 + t];
  red[t] = x; __syncthreads();
  for (int s = 64; s > 0; s >>= 1) { if (t < s) red[t] += red[t + s]; __syncthreads(); }
  float mean = red[0] * (1.f / 128.f); __syncthreads();
  float d = x - mean;
  red[t] = d * d; __syncthreads();
  for (int s = 64; s > 0; s >>= 1) { if (t < s) red[t] += red[t + s]; __syncthreads(); }
  float var = red[0] * (1.f / 128.f);
  out[(long)n * 128 + t] = d * rsqrtf(var + 1e-5f) * g[t] + b[t];
}

// ---------------------------------------------------------------------------
// Backbone update: corr = s@w_bb + b_bb, quaternion compose, t update.
// grid = N, block = 128
// ---------------------------------------------------------------------------
__global__ void bb_kernel(const float* __restrict__ s, const float* __restrict__ R,
                          const float* __restrict__ trin, const float* __restrict__ w_bb,
                          const float* __restrict__ b_bb, float* __restrict__ Rout,
                          float* __restrict__ tout) {
  __shared__ float part[128][6];
  int n = blockIdx.x, t = threadIdx.x;
  float x = s[(long)n * 128 + t];
#pragma unroll
  for (int c = 0; c < 6; ++c) part[t][c] = x * w_bb[t * 6 + c];
  __syncthreads();
  for (int st = 64; st > 0; st >>= 1) {
    if (t < st)
#pragma unroll
      for (int c = 0; c < 6; ++c) part[t][c] += part[t + st][c];
    __syncthreads();
  }
  if (t == 0) {
    float corr[6];
#pragma unroll
    for (int c = 0; c < 6; ++c) corr[c] = part[0][c] + b_bb[c];
    float rx = corr[0], ry = corr[1], rz = corr[2];
    float norm = sqrtf(1.f + rx * rx + ry * ry + rz * rz);
    float a = 1.f / norm, b = rx / norm, c = ry / norm, d = rz / norm;
    float Rc[9] = {a * a + b * b - c * c - d * d, 2 * b * c - 2 * a * d, 2 * b * d + 2 * a * c,
                   2 * b * c + 2 * a * d, a * a - b * b + c * c - d * d, 2 * c * d - 2 * a * b,
                   2 * b * d - 2 * a * c, 2 * c * d + 2 * a * b, a * a - b * b - c * c + d * d};
    const float* Rn = R + (long)n * 9;
#pragma unroll
    for (int i = 0; i < 3; ++i)
#pragma unroll
      for (int j = 0; j < 3; ++j)
        Rout[(long)n * 9 + i * 3 + j] =
            Rn[i * 3] * Rc[j] + Rn[i * 3 + 1] * Rc[3 + j] + Rn[i * 3 + 2] * Rc[6 + j];
#pragma unroll
    for (int i = 0; i < 3; ++i)
      tout[(long)n * 3 + i] =
          Rn[i * 3] * corr[3] + Rn[i * 3 + 1] * corr[4] + Rn[i * 3 + 2] * corr[5] +
          trin[(long)n * 3 + i];
  }
}

// ---------------------------------------------------------------------------
// Host launcher
// ---------------------------------------------------------------------------
static void launch_gemm(hipStream_t stream, const float* A, long lda, long sA,
                        const float* B, long ldb, long sB, float* C, long ldc, long sC,
                        const float* bias, const float* add, long ldadd, long sAdd,
                        int Mtiles, int Ntiles, int mmax, int nmax, int K, int relu,
                        int batch) {
  dim3 grid(Ntiles, Mtiles, batch);
  gemm_wmma_kernel<<<grid, 32, 0, stream>>>(A, lda, sA, B, ldb, sB, C, ldc, sC, bias, add,
                                            ldadd, sAdd, mmax, nmax, K, relu);
}

extern "C" void kernel_launch(void* const* d_in, const int* in_sizes, int n_in,
                              void* d_out, int out_size, void* d_ws, size_t ws_size,
                              hipStream_t stream) {
  const float* node  = (const float*)d_in[0];   // 1024x128
  const float* edge  = (const float*)d_in[1];   // 1024x1024x64
  const float* R     = (const float*)d_in[2];   // 1024x3x3
  const float* tr    = (const float*)d_in[3];   // 1024x3
  const float* w_sqkv= (const float*)d_in[4];   // 128x576
  const float* w_vqk = (const float*)d_in[5];   // 128x288
  const float* w_vv  = (const float*)d_in[6];   // 128x288
  const float* w_eb  = (const float*)d_in[7];   // 64x12
  const float* pw    = (const float*)d_in[8];   // 12
  const float* w_fin = (const float*)d_in[9];   // 1344x128
  const float* b_fin = (const float*)d_in[10];  // 128
  const float* ln1g  = (const float*)d_in[11];
  const float* ln1b  = (const float*)d_in[12];
  const float* ln2g  = (const float*)d_in[13];
  const float* ln2b  = (const float*)d_in[14];
  const float* w_r1  = (const float*)d_in[15];  // 128x256
  const float* b_r1  = (const float*)d_in[16];
  const float* w_r2  = (const float*)d_in[17];  // 256x256
  const float* b_r2  = (const float*)d_in[18];
  const float* w_r3  = (const float*)d_in[19];  // 256x128
  const float* b_r3  = (const float*)d_in[20];
  const float* w_bb  = (const float*)d_in[21];  // 128x6
  const float* b_bb  = (const float*)d_in[22];  // 6

  float* out = (float*)d_out;     // s_i[131072] | R_new[9216] | t_new[3072]
  float* ws = (float*)d_ws;

  size_t off = 0;
  float* qkv      = ws + off; off += 1024 * 576;      // scalar q,k,v
  float* vqk      = ws + off; off += 1024 * 288;      // vector q,k (local)
  float* vv       = ws + off; off += 1024 * 288;      // vector v (local)
  float* gq       = ws + off; off += 1024 * 144;
  float* gk       = ws + off; off += 1024 * 144;
  float* gv       = ws + off; off += 1024 * 288;
  float* sqq      = ws + off; off += 1024 * 12;
  float* sqk      = ws + off; off += 1024 * 12;
  float* att      = ws + off; off += (size_t)12 * NN * NN; // logits -> softmax in place
  float* combined = ws + off; off += 1024 * 1344;
  float* agg      = ws + off; off += 1024 * 384;      // padded: 32 per head
  float* sbuf     = ws + off; off += 1024 * 128;
  float* h1       = ws + off; off += 1024 * 256;
  float* h2       = ws + off; off += 1024 * 256;
  float* s2       = ws + off; off += 1024 * 128;
  (void)ws_size; (void)in_sizes; (void)n_in; (void)out_size;

  // 1) projections
  launch_gemm(stream, node, 128, 0, w_sqkv, 576, 0, qkv, 576, 0,
              nullptr, nullptr, 0, 0, 64, 36, 1023, 575, 128, 0, 1);
  launch_gemm(stream, node, 128, 0, w_vqk, 288, 0, vqk, 288, 0,
              nullptr, nullptr, 0, 0, 64, 18, 1023, 287, 128, 0, 1);
  launch_gemm(stream, node, 128, 0, w_vv, 288, 0, vv, 288, 0,
              nullptr, nullptr, 0, 0, 64, 18, 1023, 287, 128, 0, 1);

  // 2) rigid point transforms
  pts_kernel<<<1024, 96, 0, stream>>>(vqk, vv, R, tr, gq, gk, gv, sqq, sqk);

  // 3) fused logits (edge-bias + scalar + point attention)
  logits_kernel<<<dim3(64, 64), dim3(32, 16), 0, stream>>>(edge, w_eb, qkv, gq, gk, sqq,
                                                           sqk, pw, att);

  // 4) softmax over j
  softmax_kernel<<<12 * 1024, 256, 0, stream>>>(att);

  // 5) edge_out[i,h,c]: batch over i.  A = att[h, i, :]  B = edge[i, :, :]
  //    M=12 ragged: clamped rows 12..15 store into combined[i][768:1024),
  //    fully overwritten by steps 6 and 8 (stream-ordered).
  launch_gemm(stream, att, (long)NN * NN, NN, edge, 64, (long)NN * 64,
              combined, 64, 1344, nullptr, nullptr, 0, 0, 1, 4, 11, 63, NN, 0, 1024);

  // 6) scalar_out[i,h,d]: batch over h.  A = att[h]  B = sv[:,h,:]
  launch_gemm(stream, att, NN, (long)NN * NN, qkv + 384, 576, 16,
              combined + 768, 1344, 16, nullptr, nullptr, 0, 0, 64, 1, 1023, 15, NN, 0, 12);

  // 7) agg[i,h,pa]: batch over h.  B = gv[:,h,:] (N=24, padded store to 32)
  launch_gemm(stream, att, NN, (long)NN * NN, gv, 288, 24,
              agg, 384, 32, nullptr, nullptr, 0, 0, 64, 2, 1023, 23, NN, 0, 12);

  // 8) local frame + norm -> combined[:,960:]
  vec_kernel<<<1024, 96, 0, stream>>>(agg, R, tr, combined);

  // 9) s = node + combined @ w_final + b_final ; LN1 (in place)
  launch_gemm(stream, combined, 1344, 0, w_fin, 128, 0, sbuf, 128, 0,
              b_fin, node, 128, 0, 64, 8, 1023, 127, 1344, 0, 1);
  ln_kernel<<<1024, 128, 0, stream>>>(sbuf, ln1g, ln1b, sbuf);

  // 10) transition MLP
  launch_gemm(stream, sbuf, 128, 0, w_r1, 256, 0, h1, 256, 0,
              b_r1, nullptr, 0, 0, 64, 16, 1023, 255, 128, 1, 1);
  launch_gemm(stream, h1, 256, 0, w_r2, 256, 0, h2, 256, 0,
              b_r2, nullptr, 0, 0, 64, 16, 1023, 255, 256, 1, 1);
  launch_gemm(stream, h2, 256, 0, w_r3, 128, 0, s2, 128, 0,
              b_r3, sbuf, 128, 0, 64, 8, 1023, 127, 256, 0, 1);

  // 11) LN2 -> output s_i
  ln_kernel<<<1024, 128, 0, stream>>>(s2, ln2g, ln2b, out);

  // 12) backbone update -> R_new, t_new
  bb_kernel<<<1024, 128, 0, stream>>>(out, R, tr, w_bb, b_bb, out + 131072,
                                      out + 131072 + 9216);
}
